// HGRU4REC_64295660421643
// MI455X (gfx1250) — matrix operand order repaired and checked by reading, not compiled, not run on hardware
//
#include <hip/hip_runtime.h>
#include <hip/hip_bf16.h>

// Problem constants (match reference)
#define B_  2048
#define H_  512
#define H3_ 1536
#define V_  50000

typedef __attribute__((ext_vector_type(16))) __bf16 v16bf;
typedef __attribute__((ext_vector_type(8)))  float  v8f;

// ---------------------------------------------------------------------------
// CDNA5 async-copy helpers (inline asm; ISA §10, §11.2.4, §15.18)
// ---------------------------------------------------------------------------
#if __has_builtin(__builtin_amdgcn_s_wait_asynccnt)
#define WAIT_ASYNC(n) __builtin_amdgcn_s_wait_asynccnt(n)
#else
#define WAIT_ASYNC(n) asm volatile("s_wait_asynccnt %0" ::"i"(n) : "memory")
#endif

// DMA 16 bytes global -> LDS, tracked with ASYNCcnt (per-lane addresses).
__device__ __forceinline__ void async_copy_b128(unsigned lds_off,
                                                const __bf16* src)
{
    unsigned long long ga = (unsigned long long)(uintptr_t)src;
    asm volatile("global_load_async_to_lds_b128 %0, %1, off"
                 :: "v"(lds_off), "v"(ga) : "memory");
}

// ---------------------------------------------------------------------------
// NT GEMM: C[M,N] = act( A[M,K](bf16) @ Bw[N,K](bf16)^T + bias )
// Block = 128 threads = 4 wave32.  Block tile M=16, N=128; each wave owns a
// 16x32 C tile (2 accumulators).  B tiles are DMA'd to LDS (double-buffered,
// async), fragments fetched with plain ds_load_b128 (no transpose).
// ---------------------------------------------------------------------------
__global__ void __launch_bounds__(128)
gemm_nt_bf16_async_kernel(const __bf16* __restrict__ A,   // [M,K] bf16
                          const __bf16* __restrict__ Bw,  // [N,K] bf16
                          const float*  __restrict__ bias,// [N]
                          float*        __restrict__ C,   // [M,N]
                          int M, int N, int K, int act)
{
    __shared__ alignas(16) __bf16 Bs[2][128 * 32];   // [buf][n][k], 8KB each

    const int t    = threadIdx.x;
    const int lane = t & 31;
    const int wave = t >> 5;
    const int m0   = blockIdx.y << 4;
    const int nBlk = blockIdx.x << 7;          // block's 128-wide N origin

    // A fragment addressing (16-bit A 16x32, ISA 7.12.2): row M = lane&15,
    // lane<16 holds K {0..7,16..23}; lane>=16 holds K {8..15,24..31}.
    const int     mRow  = m0 + (lane & 15);
    const int     kHalf = (lane >> 4) << 3;    // 0 or 8
    const __bf16* pA    = A + (size_t)mRow * K;

    // Async copy plan: tile = 128 rows x 32 k of bf16 = 512 x 16B chunks;
    // each of 128 threads DMAs 4 chunks.  Rows clamped (stores are guarded).
    unsigned      dst_off[4];
    const __bf16* srcRow[4];
#pragma unroll
    for (int j = 0; j < 4; ++j) {
        int c  = (j << 7) + t;                 // 0..511
        int n  = c >> 2;                       // 0..127
        int kc = (c & 3) << 3;                 // 0,8,16,24
        int ng = nBlk + n; if (ng > N - 1) ng = N - 1;
        srcRow[j]  = Bw + (size_t)ng * K + kc; // + ks at issue time
        dst_off[j] = (unsigned)(n * 32 + kc) * 2;
    }
    const unsigned bufBase[2] = { (unsigned)(uintptr_t)&Bs[0][0],
                                  (unsigned)(uintptr_t)&Bs[1][0] };

    // ds_load_b128 lane addresses: four 16x16 bf16 subtiles
    // (n-subtile s in {0,1}) x (k-half kh in {0,1}); lane = source row + chunk.
    const int nLoc = wave << 5;                // wave's local N origin
    unsigned trb[4];
#pragma unroll
    for (int s = 0; s < 2; ++s)
#pragma unroll
        for (int kh = 0; kh < 2; ++kh)
            trb[s * 2 + kh] = (unsigned)(((nLoc + s * 16) * 32 + kh * 16) * 2)
                            + (unsigned)(lane & 15) * 64
                            + (unsigned)(lane >> 4) * 16;

    v8f acc0 = {}, acc1 = {};

    // Prologue: DMA buffer 0
#pragma unroll
    for (int j = 0; j < 4; ++j)
        async_copy_b128(bufBase[0] + dst_off[j], srcRow[j]);

    for (int ks = 0; ks < K; ks += 32) {
        const int  cur  = (ks >> 5) & 1;
        const bool more = (ks + 32) < K;
        if (more) {                            // DMA next tile into other buf
#pragma unroll
            for (int j = 0; j < 4; ++j)
                async_copy_b128(bufBase[cur ^ 1] + dst_off[j],
                                srcRow[j] + ks + 32);
            WAIT_ASYNC(4);                     // current buf's 4 DMAs done
        } else {
            WAIT_ASYNC(0);
        }
        __syncthreads();                       // all waves' DMAs visible

        union { v16bf v; uint4 q[2]; } afr, bfr0, bfr1;
        const __bf16* pAk = pA + ks + kHalf;
        afr.q[0] = *(const uint4*)(pAk);
        afr.q[1] = *(const uint4*)(pAk + 16);

        uint4 b0, b1, b2, b3;
        const unsigned base = bufBase[cur];
        asm volatile("ds_load_b128 %0, %4\n\t"
                     "ds_load_b128 %1, %5\n\t"
                     "ds_load_b128 %2, %6\n\t"
                     "ds_load_b128 %3, %7\n\t"
                     "s_wait_dscnt 0x0"
                     : "=&v"(b0), "=&v"(b1), "=&v"(b2), "=&v"(b3)
                     : "v"(base + trb[0]), "v"(base + trb[1]),
                       "v"(base + trb[2]), "v"(base + trb[3])
                     : "memory");
        bfr0.q[0] = b0; bfr0.q[1] = b1;
        bfr1.q[0] = b2; bfr1.q[1] = b3;

        acc0 = __builtin_amdgcn_wmma_f32_16x16x32_bf16(
                   false, afr.v, false, bfr0.v, (short)0, acc0, false, false);
        acc1 = __builtin_amdgcn_wmma_f32_16x16x32_bf16(
                   false, afr.v, false, bfr1.v, (short)0, acc1, false, false);
        __syncthreads();                       // reads done before buf reuse
    }

    // C layout (32-bit C 16x16): VGPR r -> row m0 + r + 8*(lane>>4)
    const int rbase = m0 + ((lane >> 4) << 3);
#pragma unroll
    for (int s = 0; s < 2; ++s) {
        const int col = nBlk + nLoc + s * 16 + (lane & 15);
        if (col < N) {
            const v8f   acc = s ? acc1 : acc0;
            const float bv  = bias ? bias[col] : 0.f;
#pragma unroll
            for (int r = 0; r < 8; ++r) {
                float x = acc[r] + bv;
                if (act) x = tanhf(x);
                C[(size_t)(rbase + r) * N + col] = x;
            }
        }
    }
}

// ---------------------------------------------------------------------------
// f32 -> bf16 bulk converter (8 elements/thread, 16B stores)
// ---------------------------------------------------------------------------
__global__ void convertw_kernel(const float* __restrict__ in,
                                __bf16* __restrict__ out, long n)
{
    long t = ((long)blockIdx.x * 256 + threadIdx.x) * 8;
    if (t + 8 > n) return;
    float4 a = *(const float4*)(in + t);
    float4 b = *(const float4*)(in + t + 4);
    union { __bf16 h[8]; uint4 q; } u;
    u.h[0] = (__bf16)a.x; u.h[1] = (__bf16)a.y;
    u.h[2] = (__bf16)a.z; u.h[3] = (__bf16)a.w;
    u.h[4] = (__bf16)b.x; u.h[5] = (__bf16)b.y;
    u.h[6] = (__bf16)b.z; u.h[7] = (__bf16)b.w;
    *(uint4*)(out + t) = u.q;
}

// x_proj_s[b, j] = Wih_s[j, inputs[b]] + bih_s[j]
__global__ void gather_xps_kernel(const float* __restrict__ Wih_s,
                                  const float* __restrict__ bih_s,
                                  const int*   __restrict__ idx,
                                  float*       __restrict__ xps)
{
    int j = blockIdx.x * 256 + threadIdx.x;
    int b = blockIdx.y;
    if (j < H3_)
        xps[(size_t)b * H3_ + j] = Wih_s[(size_t)j * V_ + idx[b]] + bih_s[j];
}

__device__ __forceinline__ float sigm(float x) { return 1.f / (1.f + expf(-x)); }

// user GRU cell + mask blends; emits f32 (output) + bf16 (GEMM feed)
__global__ void user_gru_kernel(const float* __restrict__ xp,
                                const float* __restrict__ hp,
                                const float* __restrict__ h_in,
                                const float* __restrict__ sm_,
                                const float* __restrict__ um_,
                                float*  __restrict__ out_f32,
                                __bf16* __restrict__ out_bf)
{
    int t = blockIdx.x * 256 + threadIdx.x;
    if (t >= B_ * H_) return;
    int b = t / H_;
    int h = t - b * H_;
    size_t base = (size_t)b * H3_;
    float xr = xp[base + h], xz = xp[base + H_ + h], xn = xp[base + 2 * H_ + h];
    float hr = hp[base + h], hz = hp[base + H_ + h], hn = hp[base + 2 * H_ + h];
    float r = sigm(xr + hr);
    float z = sigm(xz + hz);
    float nn = tanhf(xn + r * hn);
    float hprev = h_in[t];
    float upd = (1.f - z) * nn + z * hprev;
    float sm = sm_[b], um = um_[b];
    float out = (1.f - um) * (sm * upd + (1.f - sm) * hprev);
    out_f32[t] = out;
    out_bf[t]  = (__bf16)out;
}

// session_repr blend after u2s (si already tanh'ed by GEMM epilogue)
__global__ void u2s_blend_kernel(const float* __restrict__ si,
                                 const float* __restrict__ sess_in,
                                 const float* __restrict__ sm_,
                                 const float* __restrict__ um_,
                                 float*  __restrict__ mid_f32,
                                 __bf16* __restrict__ mid_bf)
{
    int t = blockIdx.x * 256 + threadIdx.x;
    if (t >= B_ * H_) return;
    int b = t / H_;
    float sm = sm_[b], um = um_[b];
    float v = (1.f - um) * (sm * si[t] + (1.f - sm) * sess_in[t]);
    mid_f32[t] = v;
    mid_bf[t]  = (__bf16)v;
}

// session GRU cell (no masks); emits f32 (output) + bf16 (score GEMM feed)
__global__ void sess_gru_kernel(const float* __restrict__ xp,
                                const float* __restrict__ hp,
                                const float* __restrict__ h_in,
                                float*  __restrict__ out_f32,
                                __bf16* __restrict__ out_bf)
{
    int t = blockIdx.x * 256 + threadIdx.x;
    if (t >= B_ * H_) return;
    int b = t / H_;
    int h = t - b * H_;
    size_t base = (size_t)b * H3_;
    float xr = xp[base + h], xz = xp[base + H_ + h], xn = xp[base + 2 * H_ + h];
    float hr = hp[base + h], hz = hp[base + H_ + h], hn = hp[base + 2 * H_ + h];
    float r = sigm(xr + hr);
    float z = sigm(xz + hz);
    float nn = tanhf(xn + r * hn);
    float hprev = h_in[t];
    float out = (1.f - z) * nn + z * hprev;
    out_f32[t] = out;
    out_bf[t]  = (__bf16)out;
}

// ---------------------------------------------------------------------------
// Launch
// ---------------------------------------------------------------------------
extern "C" void kernel_launch(void* const* d_in, const int* in_sizes, int n_in,
                              void* d_out, int out_size, void* d_ws, size_t ws_size,
                              hipStream_t stream)
{
    (void)in_sizes; (void)n_in; (void)out_size; (void)ws_size;

    const int*   inputs  = (const int*)  d_in[0];
    const float* sess_in = (const float*)d_in[1];
    const float* sm      = (const float*)d_in[2];
    const float* user_in = (const float*)d_in[3];
    const float* um      = (const float*)d_in[4];
    const float* Wih_s   = (const float*)d_in[5];
    const float* bih_s   = (const float*)d_in[6];
    const float* Whh_s   = (const float*)d_in[7];
    const float* bhh_s   = (const float*)d_in[8];
    const float* Wih_u   = (const float*)d_in[9];
    const float* bih_u   = (const float*)d_in[10];
    const float* Whh_u   = (const float*)d_in[11];
    const float* bhh_u   = (const float*)d_in[12];
    const float* u2s_W   = (const float*)d_in[13];
    const float* u2s_b   = (const float*)d_in[14];
    const float* s2o_W   = (const float*)d_in[15];
    const float* s2o_b   = (const float*)d_in[16];

    float* score    = (float*)d_out;              // [B,V]
    float* sess_out = score + (size_t)B_ * V_;    // [B,H]
    float* user_out = sess_out + (size_t)B_ * H_; // [B,H]

    char* w = (char*)d_ws;
    auto carve = [&](size_t bytes) {
        void* p = (void*)w;
        w += (bytes + 255) & ~(size_t)255;
        return p;
    };
    float*  xp      = (float*) carve((size_t)B_ * H3_ * 4);
    float*  hp      = (float*) carve((size_t)B_ * H3_ * 4);
    float*  si      = (float*) carve((size_t)B_ * H_ * 4);
    float*  mid_f32 = (float*) carve((size_t)B_ * H_ * 4);
    __bf16* sess_bf = (__bf16*)carve((size_t)B_ * H_ * 2);
    __bf16* user_bf = (__bf16*)carve((size_t)B_ * H_ * 2);
    __bf16* uout_bf = (__bf16*)carve((size_t)B_ * H_ * 2);
    __bf16* mid_bf  = (__bf16*)carve((size_t)B_ * H_ * 2);
    __bf16* sout_bf = (__bf16*)carve((size_t)B_ * H_ * 2);
    __bf16* WihU_bf = (__bf16*)carve((size_t)H3_ * H_ * 2);
    __bf16* WhhU_bf = (__bf16*)carve((size_t)H3_ * H_ * 2);
    __bf16* WhhS_bf = (__bf16*)carve((size_t)H3_ * H_ * 2);
    __bf16* U2S_bf  = (__bf16*)carve((size_t)H_ * H_ * 2);
    __bf16* S2O_bf  = (__bf16*)carve((size_t)V_ * H_ * 2);

    const long BH   = (long)B_ * H_;
    const long W3H  = (long)H3_ * H_;
    const long WHH  = (long)H_ * H_;
    const long W2O  = (long)V_ * H_;
    const dim3 blkE(256);
    const dim3 grdE((B_ * H_ + 255) / 256);
    const dim3 blkG(128);
    const dim3 g3H((H3_ + 127) / 128, B_ / 16);   // N=1536
    const dim3 gH ((H_  + 127) / 128, B_ / 16);   // N=512
    const dim3 gV ((V_  + 127) / 128, B_ / 16);   // N=50000
    auto cvtGrid = [](long n) { return dim3((unsigned)((n / 8 + 255) / 256)); };

    // 0) one-time bf16 conversions (weights + GEMM activations)
    convertw_kernel<<<cvtGrid(W3H), blkE, 0, stream>>>(Wih_u, WihU_bf, W3H);
    convertw_kernel<<<cvtGrid(W3H), blkE, 0, stream>>>(Whh_u, WhhU_bf, W3H);
    convertw_kernel<<<cvtGrid(W3H), blkE, 0, stream>>>(Whh_s, WhhS_bf, W3H);
    convertw_kernel<<<cvtGrid(WHH), blkE, 0, stream>>>(u2s_W, U2S_bf, WHH);
    convertw_kernel<<<cvtGrid(W2O), blkE, 0, stream>>>(s2o_W, S2O_bf, W2O);
    convertw_kernel<<<cvtGrid(BH),  blkE, 0, stream>>>(sess_in, sess_bf, BH);
    convertw_kernel<<<cvtGrid(BH),  blkE, 0, stream>>>(user_in, user_bf, BH);

    // 1) user GRU projections
    gemm_nt_bf16_async_kernel<<<g3H, blkG, 0, stream>>>(
        sess_bf, WihU_bf, bih_u, xp, B_, H3_, H_, 0);
    gemm_nt_bf16_async_kernel<<<g3H, blkG, 0, stream>>>(
        user_bf, WhhU_bf, bhh_u, hp, B_, H3_, H_, 0);
    // 2) user GRU cell + masks -> user_repr output
    user_gru_kernel<<<grdE, blkE, 0, stream>>>(xp, hp, user_in, sm, um,
                                               user_out, uout_bf);
    // 3) session init: tanh(user_out @ u2s_W^T + b), then blend
    gemm_nt_bf16_async_kernel<<<gH, blkG, 0, stream>>>(
        uout_bf, U2S_bf, u2s_b, si, B_, H_, H_, 1);
    u2s_blend_kernel<<<grdE, blkE, 0, stream>>>(si, sess_in, sm, um,
                                                mid_f32, mid_bf);
    // 4) session GRU: gathered x_proj + recurrent projection
    gather_xps_kernel<<<dim3((H3_ + 255) / 256, B_), blkE, 0, stream>>>(
        Wih_s, bih_s, inputs, xp);
    gemm_nt_bf16_async_kernel<<<g3H, blkG, 0, stream>>>(
        mid_bf, WhhS_bf, bhh_s, hp, B_, H3_, H_, 0);
    sess_gru_kernel<<<grdE, blkE, 0, stream>>>(xp, hp, mid_f32,
                                               sess_out, sout_bf);
    // 5) output scores: tanh(sess_out @ s2o_W^T + b) — the 105 GFLOP GEMM
    gemm_nt_bf16_async_kernel<<<gV, blkG, 0, stream>>>(
        sout_bf, S2O_bf, s2o_b, score, B_, V_, H_, 1);
}
